// QuantizedPatternMatcher_11768210391675
// MI455X (gfx1250) — 8-bit matrix operand order repaired and checked by reading, compiled, not run on hardware
//
#include <hip/hip_runtime.h>
#include <hip/hip_bf16.h>

typedef __attribute__((ext_vector_type(16))) int          v16i;
typedef __attribute__((ext_vector_type(8)))  float        v8f;
typedef __attribute__((ext_vector_type(8)))  int          v8i;
typedef __attribute__((ext_vector_type(4)))  int          v4i;
typedef __attribute__((ext_vector_type(4)))  unsigned int v4u;

#define D_DIM   512
#define KBYTES  4096
#define P_DIM   1024
#define M_DIM   8192
#define NTILES  (P_DIM / 16)
#define NB      4
#define MB      2
#define KC      512
#define NCHUNK  (KBYTES / KC)
#define KSTEPS  (KC / 128)

#if defined(__has_builtin)
#if __has_builtin(__builtin_amdgcn_tensor_load_to_lds) && \
    __has_builtin(__builtin_amdgcn_s_wait_tensorcnt)
#define USE_TDM 1
#endif
#endif
#ifndef USE_TDM
#define USE_TDM 0
#endif

__global__ __launch_bounds__(256)
void qpm_quant_onehot(const float* __restrict__ x,
                      const float* __restrict__ edges,
                      unsigned long long* __restrict__ q64,
                      int n)
{
    int i = blockIdx.x * 256 + threadIdx.x;
    if (i >= n) return;
    float v = x[i];
    int bin = 0;
#pragma unroll
    for (int e = 0; e < 7; ++e) bin += (v > edges[e]) ? 1 : 0;
    q64[i] = 0x38ull << (bin * 8);
}

#if USE_TDM
__device__ __forceinline__ void tdm_issue_panel(unsigned lds_off,
                                                const unsigned char* gptr,
                                                int kbyte_off)
{
    unsigned long long ga = (unsigned long long)(size_t)gptr +
                            (unsigned long long)(unsigned)kbyte_off;
    v4u g0;
    g0[0] = 1u;
    g0[1] = lds_off;
    g0[2] = (unsigned)(ga & 0xFFFFFFFFu);
    g0[3] = (unsigned)((ga >> 32) & 0x1FFFFFFu)
            | (2u << 30);
    v8i g1;
    g1[0] = 0;
    g1[1] = (int)(4096u << 16);
    g1[2] = (int)(1024u << 16);
    g1[3] = (int)((unsigned)KC << 16);
    g1[4] = 64;
    g1[5] = 4096;
    g1[6] = 0;
    g1[7] = 0;
    v4i gz4 = {0, 0, 0, 0};
    v8i gz8 = {0, 0, 0, 0, 0, 0, 0, 0};
    __builtin_amdgcn_tensor_load_to_lds(g0, g1, gz4, gz4, gz8, 0);
}
#endif

#define CONSUME_CHUNK(BUF, CHUNK)                                              \
    do {                                                                       \
        _Pragma("unroll 1")                                                    \
        for (int ks = 0; ks < KSTEPS; ++ks) {                                  \
            const int kg = (CHUNK) * KC + ks * 128;                            \
            v16i afrag[MB];                                                    \
            _Pragma("unroll")                                                  \
            for (int mb = 0; mb < MB; ++mb) {                                  \
                _Pragma("unroll")                                              \
                for (int v = 0; v < 16; ++v) {                                 \
                    const int koff = ((v >> 3) * 64) + (((v >> 1) & 3) * 16)   \
                                     + (half * 8) + ((v & 1) * 4);             \
                    afrag[mb][v] = *(const int*)(arow[mb] + kg + koff);        \
                }                                                              \
            }                                                                  \
            _Pragma("unroll")                                                  \
            for (int j = 0; j < NB; ++j) {                                     \
                const unsigned char* bbase =                                   \
                    &(BUF)[(j * 16 + colLane) * KC + ks * 128];                \
                v16i bfrag;                                                    \
                _Pragma("unroll")                                              \
                for (int v = 0; v < 16; ++v) {                                 \
                    const int koff = ((v >> 3) * 64) + (((v >> 1) & 3) * 16)   \
                                     + (half * 8) + ((v & 1) * 4);             \
                    bfrag[v] = *(const int*)(bbase + koff);                    \
                }                                                              \
                _Pragma("unroll")                                              \
                for (int mb = 0; mb < MB; ++mb)                                \
                    acc[mb][j] = __builtin_amdgcn_wmma_f32_16x16x128_fp8_fp8(  \
                        afrag[mb], bfrag, (short)0, acc[mb][j],                \
                         false,  false);                 \
            }                                                                  \
        }                                                                      \
    } while (0)

__global__ __launch_bounds__(256)
void qpm_wmma_match(const unsigned char* __restrict__ qx,
                    const unsigned char* __restrict__ qp,
                    float* __restrict__ out)
{
    __shared__ alignas(16) unsigned char ldsB[2][64 * KC];

    const int lane    = threadIdx.x & 31;
    const int wave    = threadIdx.x >> 5;
    const int gwave   = blockIdx.x * 8 + wave;
    const int m0      = gwave * (MB * 16);
    const int half    = lane >> 4;
    const int colLane = lane & 15;

    const unsigned char* arow[MB];
    arow[0] = qx + (size_t)(m0 + colLane) * KBYTES;
    arow[1] = arow[0] + (size_t)16 * KBYTES;

    float bestVal[MB][8];
    int   bestIdx[MB][8];
#pragma unroll
    for (int mb = 0; mb < MB; ++mb)
#pragma unroll
        for (int v = 0; v < 8; ++v) { bestVal[mb][v] = -1.0f; bestIdx[mb][v] = 0; }

#pragma unroll 1
    for (int nb = 0; nb < NTILES; nb += NB) {
        v8f acc[MB][NB];
#pragma unroll
        for (int mb = 0; mb < MB; ++mb)
#pragma unroll
            for (int j = 0; j < NB; ++j)
                acc[mb][j] = (v8f){0.f, 0.f, 0.f, 0.f, 0.f, 0.f, 0.f, 0.f};

        const unsigned char* qpPanel = qp + (size_t)(nb * 16) * KBYTES;

#if USE_TDM
        if (wave == 0) {
            tdm_issue_panel((unsigned)(size_t)&ldsB[0][0], qpPanel, 0);
            __builtin_amdgcn_s_wait_tensorcnt(0);
        }
        __syncthreads();
#pragma unroll 1
        for (int c = 0; c < NCHUNK; ++c) {
            if (wave == 0 && (c + 1) < NCHUNK)
                tdm_issue_panel((unsigned)(size_t)&ldsB[(c + 1) & 1][0],
                                qpPanel, (c + 1) * KC);
            CONSUME_CHUNK(ldsB[c & 1], c);
            if (wave == 0) __builtin_amdgcn_s_wait_tensorcnt(0);
            __syncthreads();
        }
#else
#pragma unroll 1
        for (int c = 0; c < NCHUNK; ++c) {
            for (int t = threadIdx.x; t < (64 * KC) / 16; t += 256) {
                const int o = t * 16;
                const int row = o / KC, within = o % KC;
                *(v4u*)&ldsB[0][o] =
                    *(const v4u*)(qpPanel + (size_t)row * KBYTES + c * KC + within);
            }
            __syncthreads();
            CONSUME_CHUNK(ldsB[0], c);
            __syncthreads();
        }
#endif

#pragma unroll
        for (int j = 0; j < NB; ++j) {
            const int col = (nb + j) * 16 + colLane;
#pragma unroll
            for (int mb = 0; mb < MB; ++mb)
#pragma unroll
                for (int v = 0; v < 8; ++v) {
                    float c = acc[mb][j][v];
                    if (c > bestVal[mb][v]) { bestVal[mb][v] = c; bestIdx[mb][v] = col; }
                }
        }
    }

#pragma unroll
    for (int mask = 1; mask < 16; mask <<= 1) {
#pragma unroll
        for (int mb = 0; mb < MB; ++mb)
#pragma unroll
            for (int v = 0; v < 8; ++v) {
                float oc = __shfl_xor(bestVal[mb][v], mask, 32);
                int   oi = __shfl_xor(bestIdx[mb][v], mask, 32);
                if (oc > bestVal[mb][v] ||
                    (oc == bestVal[mb][v] && oi < bestIdx[mb][v])) {
                    bestVal[mb][v] = oc;
                    bestIdx[mb][v] = oi;
                }
            }
    }

    if (colLane == 0) {
#pragma unroll
        for (int mb = 0; mb < MB; ++mb)
#pragma unroll
            for (int v = 0; v < 8; ++v) {
                const int m = m0 + mb * 16 + half * 8 + v;
                out[m]         = (float)bestIdx[mb][v];
                out[M_DIM + m] = bestVal[mb][v] * (1.0f / (float)D_DIM);
            }
    }
}

extern "C" void kernel_launch(void* const* d_in, const int* in_sizes, int n_in,
                              void* d_out, int out_size, void* d_ws, size_t ws_size,
                              hipStream_t stream)
{
    const float* x        = (const float*)d_in[0];
    const float* patterns = (const float*)d_in[1];
    const float* edges    = (const float*)d_in[2];
    float*       out      = (float*)d_out;

    unsigned char* qx = (unsigned char*)d_ws;
    unsigned char* qp = qx + (size_t)M_DIM * KBYTES;

    const int nx = M_DIM * D_DIM;
    const int np = P_DIM * D_DIM;

    qpm_quant_onehot<<<(nx + 255) / 256, 256, 0, stream>>>(
        x, edges, (unsigned long long*)qx, nx);
    qpm_quant_onehot<<<(np + 255) / 256, 256, 0, stream>>>(
        patterns, edges, (unsigned long long*)qp, np);

    qpm_wmma_match<<<32, 256, 0, stream>>>(qx, qp, out);
}
